// Wavelet_model_68341519613957
// MI455X (gfx1250) — matrix operand order repaired and checked by reading, not compiled, not run on hardware
//
#include <hip/hip_runtime.h>
#include <hip/hip_bf16.h>

// ---------------------------------------------------------------------------
// Shapes from the reference
// ---------------------------------------------------------------------------
#define Bc     4
#define Ac     128
#define Lc     101
#define Ec     20
#define Pc     10
#define PDc    3
#define DEPTHc 4
#define Hc     64
#define NBAL   (Bc*Ac*Lc)          // 51712
// Padded tile dims (multiples of 16 for WMMA)
#define LP 128
#define EP 32
#define HP 64
#define SCALE_F 0.22360679774997896f   // E^{-0.5} = 1/sqrt(20)

typedef __attribute__((ext_vector_type(16))) _Float16 v16h;
typedef __attribute__((ext_vector_type(8)))  float    v8f;

// ---------------------------------------------------------------------------
// Workgroup GEMM using V_WMMA_F32_16X16X32_F16 (wave32, f32 accumulate).
//   C[M][N](f16,LDS) = act( (A[M][K](f16,LDS) x B + bias) * scale )
// B source:
//   B_GWT: global f32 weight, row-major [Nreal][Kreal] (i.e. x @ W^T), padded
//          with zeros via bounds checks (only for the two 101x101 weights).
//   B_LWT: LDS f16, row-major [N][K], already zero padded (weights staged once
//          per block, and K-matrix for Q.K^T) -> contiguous b128-class loads.
//   B_LKN: LDS f16, row-major [K][N]  (e.g. V for P.V)
// A fragment uses contiguous k, the same order as B:
//   lane m = lane&15, K = (lane>>4)*16 + e
// B fragment: lane n = lane&15, K = (lane>>4)*16 + e
// C store:    lane n = lane&15, rows m = (lane>>4)*8 + r
// ---------------------------------------------------------------------------
enum { B_GWT = 0, B_LWT = 1, B_LKN = 2 };

template<int BSRC, bool TRANS>
__device__ __forceinline__ void wg_gemm(
    const _Float16* A, int lda,
    const void* Bp, int ldb, int Kreal, int Nreal,
    const float* bias, bool do_relu, float scale,
    _Float16* C, int ldc, int M, int N, int K)
{
  const int lane = threadIdx.x & 31;
  const int wave = threadIdx.x >> 5;
  const int nw   = blockDim.x >> 5;
  const int nt   = N >> 4;
  const int tiles = (M >> 4) * nt;
  const int ln15 = lane & 15;
  const int hi   = lane >> 4;

  for (int t = wave; t < tiles; t += nw) {
    const int tm = (t / nt) << 4;
    const int tn = (t % nt) << 4;
    v8f c;
    {
      float bv = 0.f;
      const int cn = tn + ln15;
      if (bias && cn < Nreal) bv = bias[cn];
#pragma unroll
      for (int r = 0; r < 8; ++r) c[r] = bv;
    }
    for (int k0 = 0; k0 < K; k0 += 32) {
      v16h a;
      {
        const _Float16* Ar = A + (tm + ln15) * lda + k0 + hi * 16;
#pragma unroll
        for (int e = 0; e < 16; ++e)
          a[e] = Ar[e];
      }
      v16h b;
      {
        const int n  = tn + ln15;
        const int kk = k0 + (hi << 4);
        if (BSRC == B_GWT) {
          const float* W = (const float*)Bp;
#pragma unroll
          for (int e = 0; e < 16; ++e) {
            const int k = kk + e;
            b[e] = (_Float16)((n < Nreal && k < Kreal) ? W[n * ldb + k] : 0.f);
          }
        } else if (BSRC == B_LWT) {
          const _Float16* W = (const _Float16*)Bp + n * ldb + kk;
#pragma unroll
          for (int e = 0; e < 16; ++e) b[e] = W[e];
        } else {
          const _Float16* Bm = (const _Float16*)Bp;
#pragma unroll
          for (int e = 0; e < 16; ++e) b[e] = Bm[(kk + e) * ldb + n];
        }
      }
      c = __builtin_amdgcn_wmma_f32_16x16x32_f16(false, a, false, b,
                                                 (short)0, c, false, false);
    }
    const int n  = tn + ln15;
    const int m0 = tm + (hi << 3);
#pragma unroll
    for (int r = 0; r < 8; ++r) {
      float v = c[r] * scale;
      if (do_relu) v = fmaxf(v, 0.f);
      if (TRANS) C[n * ldc + (m0 + r)] = (_Float16)v;
      else       C[(m0 + r) * ldc + n] = (_Float16)v;
    }
  }
}

// Stage a global f32 weight [Nreal][Kreal] into padded f16 LDS [Nrows][Kcols].
__device__ __forceinline__ void stage_w(const float* __restrict__ W,
                                        int Nreal, int Kreal,
                                        _Float16* dst, int Nrows, int Kcols)
{
  for (int i = threadIdx.x; i < Nrows * Kcols; i += blockDim.x) {
    const int n = i / Kcols, k = i % Kcols;
    dst[i] = (_Float16)((n < Nreal && k < Kreal) ? W[n * Kreal + k] : 0.f);
  }
}

// ---------------------------------------------------------------------------
// Per-(d,depth) block parameters
// ---------------------------------------------------------------------------
struct BlkP {
  const float *ln_g, *ln_b, *bn_g, *bn_b, *Wlay, *blay, *Wbat, *bbat;
  const float *m1_W1, *m1_b1, *m1_W2, *m1_b2, *m1_W3, *m1_b3;
  const float *m2_Wq, *m2_bq, *m2_Wk, *m2_bk, *m2_Wv, *m2_bv, *m2_Wo, *m2_bo;
  const float *m3_Wq, *m3_bq, *m3_Wk, *m3_bk, *m3_Wv, *m3_bv, *m3_Wo, *m3_bo;
  const float *Wae, *bae, *Wan, *ban, *expl;
};

// ---------------------------------------------------------------------------
// One workgroup per (b,a) tile: full _emb_update (12 branches, softmax mix).
// Reads x (f32) from `in`, writes the weighted branch sum to `out`.
// ---------------------------------------------------------------------------
__global__ __launch_bounds__(256)
void emb_update_kernel(const float* __restrict__ in, float* __restrict__ out, BlkP bp)
{
  __shared__ _Float16 s_x16[LP][EP];
  __shared__ _Float16 s_xl [LP][EP];
  __shared__ _Float16 s_xb [LP][EP];
  __shared__ _Float16 s_xbt[EP][LP];
  __shared__ _Float16 s_t0 [LP][HP];
  __shared__ _Float16 s_t1 [LP][HP];
  __shared__ _Float16 s_t2 [LP][EP];
  __shared__ _Float16 s_t3 [LP][EP];
  __shared__ _Float16 s_P  [LP][LP];
  // staged weights (zero padded, f16)
  __shared__ _Float16 w_lay[EP][EP];     // Wlay
  __shared__ _Float16 w_m1a[HP][EP];     // m1_W1
  __shared__ _Float16 w_m1b[HP][HP];     // m1_W2
  __shared__ _Float16 w_m1c[EP][HP];     // m1_W3
  __shared__ _Float16 w_ee[9][EP][EP];   // m2 q,k,v,o | m3 q,k,v,o | Wae
  __shared__ float    s_w[12];

  const int tid = threadIdx.x;
  const float* xin  = in  + (size_t)blockIdx.x * Lc * Ec;
  float*       xout = out + (size_t)blockIdx.x * Lc * Ec;

  // warm L2/L0 for the two weights we will still read fragment-wise from global
  __builtin_prefetch(bp.Wbat, 0, 3);
  __builtin_prefetch(bp.Wan, 0, 3);

  // softmax(expl) branch weights
  if (tid == 0) {
    float mx = -1e30f;
    for (int i = 0; i < 12; ++i) mx = fmaxf(mx, bp.expl[i]);
    float sum = 0.f;
    for (int i = 0; i < 12; ++i) { float e = __expf(bp.expl[i] - mx); s_w[i] = e; sum += e; }
    for (int i = 0; i < 12; ++i) s_w[i] /= sum;
  }
  // stage the small, heavily reused weights into LDS f16 (once per block)
  stage_w(bp.Wlay,  Ec, Ec, &w_lay[0][0],  EP, EP);
  stage_w(bp.m1_W1, Hc, Ec, &w_m1a[0][0],  HP, EP);
  stage_w(bp.m1_W2, Hc, Hc, &w_m1b[0][0],  HP, HP);
  stage_w(bp.m1_W3, Ec, Hc, &w_m1c[0][0],  EP, HP);
  stage_w(bp.m2_Wq, Ec, Ec, &w_ee[0][0][0], EP, EP);
  stage_w(bp.m2_Wk, Ec, Ec, &w_ee[1][0][0], EP, EP);
  stage_w(bp.m2_Wv, Ec, Ec, &w_ee[2][0][0], EP, EP);
  stage_w(bp.m2_Wo, Ec, Ec, &w_ee[3][0][0], EP, EP);
  stage_w(bp.m3_Wq, Ec, Ec, &w_ee[4][0][0], EP, EP);
  stage_w(bp.m3_Wk, Ec, Ec, &w_ee[5][0][0], EP, EP);
  stage_w(bp.m3_Wv, Ec, Ec, &w_ee[6][0][0], EP, EP);
  stage_w(bp.m3_Wo, Ec, Ec, &w_ee[7][0][0], EP, EP);
  stage_w(bp.Wae,   Ec, Ec, &w_ee[8][0][0], EP, EP);
  // load x (f16, zero padded)
  for (int i = tid; i < LP * EP; i += blockDim.x) {
    const int l = i / EP, e = i % EP;
    s_x16[l][e] = (_Float16)((l < Lc && e < Ec) ? xin[l * Ec + e] : 0.f);
  }
  __syncthreads();

  // branch 0: identity (f32)
  for (int i = tid; i < Lc * Ec; i += blockDim.x) xout[i] = s_w[0] * xin[i];

  // ---- xl = relu( LN_E(x) @ Wlay^T + blay ) ----
  if (tid < LP) {
    const int l = tid;
    if (l < Lc) {
      float m = 0.f;
      for (int e = 0; e < Ec; ++e) m += xin[l * Ec + e];
      m /= Ec;
      float v = 0.f;
      for (int e = 0; e < Ec; ++e) { float d = xin[l * Ec + e] - m; v += d * d; }
      const float inv = rsqrtf(v / Ec + 1e-5f);
      for (int e = 0; e < EP; ++e)
        s_t2[l][e] = (e < Ec)
          ? (_Float16)((xin[l * Ec + e] - m) * inv * bp.ln_g[e] + bp.ln_b[e])
          : (_Float16)0.f;
    } else {
      for (int e = 0; e < EP; ++e) s_t2[l][e] = (_Float16)0.f;
    }
  }
  __syncthreads();
  wg_gemm<B_LWT,false>(&s_t2[0][0], EP, &w_lay[0][0], EP, Ec, Ec, bp.blay, true, 1.f,
                       &s_xl[0][0], EP, LP, EP, EP);
  __syncthreads();

  // ---- xb = swap( relu( LN_L(x^T) @ Wbat^T + bbat ) ) ----
  if (tid < EP) {
    const int c = tid;
    if (c < Ec) {
      float m = 0.f;
      for (int l = 0; l < Lc; ++l) m += xin[l * Ec + c];
      m /= Lc;
      float v = 0.f;
      for (int l = 0; l < Lc; ++l) { float d = xin[l * Ec + c] - m; v += d * d; }
      const float inv = rsqrtf(v / Lc + 1e-5f);
      for (int l = 0; l < LP; ++l)
        s_xbt[c][l] = (l < Lc)
          ? (_Float16)((xin[l * Ec + c] - m) * inv * bp.bn_g[l] + bp.bn_b[l])
          : (_Float16)0.f;
    } else {
      for (int l = 0; l < LP; ++l) s_xbt[c][l] = (_Float16)0.f;
    }
  }
  __syncthreads();
  wg_gemm<B_GWT,true>(&s_xbt[0][0], LP, bp.Wbat, Lc, Lc, Lc, bp.bbat, true, 1.f,
                      &s_xb[0][0], EP, EP, LP, LP);
  __syncthreads();

  // ---- branches 1..9: m1/m2/m3 applied to {x, xl, xb} ----
  for (int si = 0; si < 3; ++si) {
    const _Float16* src = (si == 0) ? &s_x16[0][0] : (si == 1) ? &s_xl[0][0] : &s_xb[0][0];

    // m1: E->64->64->E MLP
    wg_gemm<B_LWT,false>(src, EP, &w_m1a[0][0], EP, Ec, Hc, bp.m1_b1, true, 1.f,
                         &s_t0[0][0], HP, LP, HP, EP);
    __syncthreads();
    wg_gemm<B_LWT,false>(&s_t0[0][0], HP, &w_m1b[0][0], HP, Hc, Hc, bp.m1_b2, true, 1.f,
                         &s_t1[0][0], HP, LP, HP, HP);
    __syncthreads();
    wg_gemm<B_LWT,false>(&s_t1[0][0], HP, &w_m1c[0][0], HP, Hc, Ec, bp.m1_b3, false, 1.f,
                         &s_t3[0][0], EP, LP, EP, HP);
    __syncthreads();
    for (int i = tid; i < Lc * Ec; i += blockDim.x)
      xout[i] += s_w[1 + si] * (float)s_t3[i / Ec][i % Ec];
    __syncthreads();

    // m2: full LxL attention
    wg_gemm<B_LWT,false>(src, EP, &w_ee[0][0][0], EP, Ec, Ec, bp.m2_bq, false, 1.f,
                         &s_t0[0][0], HP, LP, EP, EP);
    wg_gemm<B_LWT,false>(src, EP, &w_ee[1][0][0], EP, Ec, Ec, bp.m2_bk, false, 1.f,
                         &s_t1[0][0], HP, LP, EP, EP);
    wg_gemm<B_LWT,false>(src, EP, &w_ee[2][0][0], EP, Ec, Ec, bp.m2_bv, false, 1.f,
                         &s_t2[0][0], EP, LP, EP, EP);
    __syncthreads();
    wg_gemm<B_LWT,false>(&s_t0[0][0], HP, &s_t1[0][0], HP, Ec, LP, nullptr, false,
                         SCALE_F, &s_P[0][0], LP, LP, LP, EP);
    __syncthreads();
    if (tid < LP) {                 // row softmax over real L
      const int l = tid;
      if (l < Lc) {
        float mx = -1e30f;
        for (int m = 0; m < Lc; ++m) mx = fmaxf(mx, (float)s_P[l][m]);
        float sum = 0.f;
        for (int m = 0; m < Lc; ++m) sum += __expf((float)s_P[l][m] - mx);
        const float inv = 1.f / sum;
        for (int m = 0; m < LP; ++m)
          s_P[l][m] = (m < Lc) ? (_Float16)(__expf((float)s_P[l][m] - mx) * inv)
                               : (_Float16)0.f;
      } else {
        for (int m = 0; m < LP; ++m) s_P[l][m] = (_Float16)0.f;
      }
    }
    __syncthreads();
    wg_gemm<B_LKN,false>(&s_P[0][0], LP, &s_t2[0][0], EP, LP, EP, nullptr, false, 1.f,
                         &s_t3[0][0], EP, LP, EP, LP);
    __syncthreads();
    wg_gemm<B_LWT,false>(&s_t3[0][0], EP, &w_ee[3][0][0], EP, Ec, Ec, bp.m2_bo, false, 1.f,
                         &s_t0[0][0], HP, LP, EP, EP);
    __syncthreads();
    for (int i = tid; i < Lc * Ec; i += blockDim.x)
      xout[i] += s_w[4 + si] * (float)s_t0[i / Ec][i % Ec];
    __syncthreads();

    // m3: per-position ExE outer-product attention
    wg_gemm<B_LWT,false>(src, EP, &w_ee[4][0][0], EP, Ec, Ec, bp.m3_bq, false, 1.f,
                         &s_t0[0][0], HP, LP, EP, EP);
    wg_gemm<B_LWT,false>(src, EP, &w_ee[5][0][0], EP, Ec, Ec, bp.m3_bk, false, 1.f,
                         &s_t1[0][0], HP, LP, EP, EP);
    wg_gemm<B_LWT,false>(src, EP, &w_ee[6][0][0], EP, Ec, Ec, bp.m3_bv, false, 1.f,
                         &s_t2[0][0], EP, LP, EP, EP);
    __syncthreads();
    if (tid < LP) {
      const int l = tid;
      if (l < Lc) {
        float qv[Ec], kv[Ec], vv[Ec];
        for (int e = 0; e < Ec; ++e) {
          qv[e] = (float)s_t0[l][e]; kv[e] = (float)s_t1[l][e]; vv[e] = (float)s_t2[l][e];
        }
        for (int i2 = 0; i2 < EP; ++i2) {
          if (i2 < Ec) {
            float mx = -1e30f;
            for (int j = 0; j < Ec; ++j) mx = fmaxf(mx, qv[i2] * kv[j] * SCALE_F);
            float sum = 0.f, o = 0.f;
            for (int j = 0; j < Ec; ++j) {
              const float e2 = __expf(qv[i2] * kv[j] * SCALE_F - mx);
              sum += e2; o += e2 * vv[j];
            }
            s_t3[l][i2] = (_Float16)(o / sum);
          } else s_t3[l][i2] = (_Float16)0.f;
        }
      } else {
        for (int i2 = 0; i2 < EP; ++i2) s_t3[l][i2] = (_Float16)0.f;
      }
    }
    __syncthreads();
    wg_gemm<B_LWT,false>(&s_t3[0][0], EP, &w_ee[7][0][0], EP, Ec, Ec, bp.m3_bo, false, 1.f,
                         &s_t0[0][0], HP, LP, EP, EP);
    __syncthreads();
    for (int i = tid; i < Lc * Ec; i += blockDim.x)
      xout[i] += s_w[7 + si] * (float)s_t0[i / Ec][i % Ec];
    __syncthreads();
  }

  // ---- branch 10: x * ( softmax_E(xl) @ Wae^T + bae ) ----
  if (tid < LP) {
    const int l = tid;
    if (l < Lc) {
      float mx = -1e30f;
      for (int e = 0; e < Ec; ++e) mx = fmaxf(mx, (float)s_xl[l][e]);
      float sum = 0.f;
      for (int e = 0; e < Ec; ++e) sum += __expf((float)s_xl[l][e] - mx);
      const float inv = 1.f / sum;
      for (int e = 0; e < EP; ++e)
        s_t3[l][e] = (e < Ec) ? (_Float16)(__expf((float)s_xl[l][e] - mx) * inv)
                              : (_Float16)0.f;
    } else {
      for (int e = 0; e < EP; ++e) s_t3[l][e] = (_Float16)0.f;
    }
  }
  __syncthreads();
  wg_gemm<B_LWT,false>(&s_t3[0][0], EP, &w_ee[8][0][0], EP, Ec, Ec, bp.bae, false, 1.f,
                       &s_t0[0][0], HP, LP, EP, EP);
  __syncthreads();
  for (int i = tid; i < Lc * Ec; i += blockDim.x)
    xout[i] += s_w[10] * (float)s_t0[i / Ec][i % Ec] * xin[i];
  __syncthreads();

  // ---- branch 11: x * swap( softmax_L(xb)^T @ Wan^T + ban ) ----
  if (tid < EP) {
    const int c = tid;
    if (c < Ec) {
      float mx = -1e30f;
      for (int l = 0; l < Lc; ++l) mx = fmaxf(mx, (float)s_xb[l][c]);
      float sum = 0.f;
      for (int l = 0; l < Lc; ++l) sum += __expf((float)s_xb[l][c] - mx);
      const float inv = 1.f / sum;
      for (int l = 0; l < LP; ++l)
        s_xbt[c][l] = (l < Lc) ? (_Float16)(__expf((float)s_xb[l][c] - mx) * inv)
                               : (_Float16)0.f;
    } else {
      for (int l = 0; l < LP; ++l) s_xbt[c][l] = (_Float16)0.f;
    }
  }
  __syncthreads();
  wg_gemm<B_GWT,true>(&s_xbt[0][0], LP, bp.Wan, Lc, Lc, Lc, bp.ban, false, 1.f,
                      &s_t3[0][0], EP, EP, LP, LP);
  __syncthreads();
  for (int i = tid; i < Lc * Ec; i += blockDim.x)
    xout[i] += s_w[11] * (float)s_t3[i / Ec][i % Ec] * xin[i];
}

// ---------------------------------------------------------------------------
// Prep: radial features (cos of linear on repeated r), mask, idx, init emb.
// ---------------------------------------------------------------------------
__global__ void prep_kernel(const float* __restrict__ x,
                            const float* __restrict__ emb_table,
                            const float* __restrict__ rr_W,
                            const float* __restrict__ rr_b,
                            float* __restrict__ emb, float* __restrict__ rbuf,
                            float* __restrict__ maskb, int* __restrict__ idxb)
{
  const int i = blockIdx.x * blockDim.x + threadIdx.x;
  if (i >= NBAL) return;
  const int l = i % Lc;
  const int a = (i / Lc) % Ac;
  const int b = i / (Lc * Ac);
  const float* row = x + ((((size_t)b * (Ac + 1) + (a + 1)) * Lc) + l) * 2;
  const float idxf = row[0];
  const float rv   = (l == 0) ? 0.f : row[1];
  const float m    = (idxf != 0.f) ? 1.f : 0.f;
  idxb[i]  = (int)(idxf * m);
  maskb[i] = m;
  // repeat(r,P) @ rr_W^T + rr_b  ==  r * rowsum(rr_W) + rr_b ; then cos, mask
  for (int p = 0; p < Pc; ++p) {
    float ws = 0.f;
    for (int j = 0; j < Pc; ++j) ws += rr_W[p * Pc + j];
    rbuf[(size_t)i * Pc + p] = cosf(rv * ws + rr_b[p]) * m;
  }
  float* er = emb + (size_t)i * Ec;
  if (l == 0) {
    const float* crow = x + (((size_t)b * (Ac + 1) + (a + 1)) * Lc) * 2;
    const int type = (int)crow[1];
    for (int e = 0; e < Ec; ++e) er[e] = emb_table[type * Ec + e] * m;
  } else {
    for (int e = 0; e < Ec; ++e) er[e] = 0.f;
  }
}

// Snapshot emb[:, :, 0, :] so the gather can run in place.
__global__ void snapshot_kernel(const float* __restrict__ emb, float* __restrict__ cent)
{
  const int i = blockIdx.x * blockDim.x + threadIdx.x;
  if (i >= Bc * Ac * Ec) return;
  const int e = i % Ec;
  const int ba = i / Ec;
  cent[i] = emb[((size_t)ba * Lc) * Ec + e];
}

// emb[b,a,l,:] = cent[b, idx[b,a,l], :] * (r . poly_w[d]) + poly_b[d]
__global__ void gather_kernel(float* __restrict__ emb, const float* __restrict__ cent,
                              const float* __restrict__ rbuf, const int* __restrict__ idxb,
                              const float* __restrict__ poly_w, const float* __restrict__ poly_b,
                              int d)
{
  const int i = blockIdx.x * blockDim.x + threadIdx.x;
  if (i >= NBAL) return;
  const int b = i / (Ac * Lc);
  float w = 0.f;
  for (int p = 0; p < Pc; ++p) w += rbuf[(size_t)i * Pc + p] * poly_w[d * Pc + p];
  const float pb = poly_b[d];
  const float* c = cent + ((size_t)b * Ac + idxb[i]) * Ec;
  float* er = emb + (size_t)i * Ec;
  for (int e = 0; e < Ec; ++e) er[e] = c[e] * w + pb;
}

// Head: mask, E->32 relu, 32->1
__global__ void head_kernel(const float* __restrict__ emb, const float* __restrict__ maskb,
                            const float* __restrict__ W1, const float* __restrict__ b1,
                            const float* __restrict__ W2, const float* __restrict__ b2,
                            float* __restrict__ out)
{
  const int i = blockIdx.x * blockDim.x + threadIdx.x;
  if (i >= NBAL) return;
  const float m = maskb[i];
  float ev[Ec];
  for (int e = 0; e < Ec; ++e) ev[e] = emb[(size_t)i * Ec + e] * m;
  float o = b2[0];
  for (int h = 0; h < 32; ++h) {
    float a = b1[h];
    for (int e = 0; e < Ec; ++e) a += W1[h * Ec + e] * ev[e];
    o += W2[h] * fmaxf(a, 0.f);
  }
  out[i] = o;
}

// ---------------------------------------------------------------------------
// Host side. Input order = setup_inputs() dict order, depth-first:
//  0:x 1:emb_table 2:rr_W 3:rr_b 4:poly_w 5:poly_b
//  6..40: blocks {ln_g,ln_b,bn_g,bn_b,Wlay,blay,Wbat,bbat,m1_W1,m1_b1,m1_W2,
//                 m1_b2,m1_W3,m1_b3,m2_Wq,m2_bq,m2_Wk,m2_bk,m2_Wv,m2_bv,
//                 m2_Wo,m2_bo,m3_Wq,m3_bq,m3_Wk,m3_bk,m3_Wv,m3_bv,m3_Wo,
//                 m3_bo,Wae,bae,Wan,ban,expl}
//  41:head_W1 42:head_b1 43:head_W2 44:head_b2
// ---------------------------------------------------------------------------
extern "C" void kernel_launch(void* const* d_in, const int* in_sizes, int n_in,
                              void* d_out, int out_size, void* d_ws, size_t ws_size,
                              hipStream_t stream)
{
  (void)in_sizes; (void)n_in; (void)out_size; (void)ws_size;
  auto F = [&](int i) { return (const float*)d_in[i]; };

  const float* x         = F(0);
  const float* emb_table = F(1);
  const float* rr_W      = F(2);
  const float* rr_b      = F(3);
  const float* poly_w    = F(4);
  const float* poly_b    = F(5);

  float* ws    = (float*)d_ws;
  float* embA  = ws;
  float* embB  = embA + (size_t)NBAL * Ec;
  float* rbuf  = embB + (size_t)NBAL * Ec;
  float* maskb = rbuf + (size_t)NBAL * Pc;
  int*   idxb  = (int*)(maskb + NBAL);
  float* cent  = (float*)(idxb + NBAL);

  const int blks = (NBAL + 255) / 256;
  prep_kernel<<<blks, 256, 0, stream>>>(x, emb_table, rr_W, rr_b,
                                        embA, rbuf, maskb, idxb);
  float* cur = embA;
  float* alt = embB;
  for (int d = 0; d < PDc; ++d) {
    snapshot_kernel<<<(Bc * Ac * Ec + 255) / 256, 256, 0, stream>>>(cur, cent);
    gather_kernel<<<blks, 256, 0, stream>>>(cur, cent, rbuf, idxb, poly_w, poly_b, d);
    for (int dep = 0; dep < DEPTHc; ++dep) {
      const int bi = d * DEPTHc + dep;
      BlkP bp;
      bp.ln_g  = F(6)  + (size_t)bi * Ec;       bp.ln_b  = F(7)  + (size_t)bi * Ec;
      bp.bn_g  = F(8)  + (size_t)bi * Lc;       bp.bn_b  = F(9)  + (size_t)bi * Lc;
      bp.Wlay  = F(10) + (size_t)bi * Ec * Ec;  bp.blay  = F(11) + (size_t)bi * Ec;
      bp.Wbat  = F(12) + (size_t)bi * Lc * Lc;  bp.bbat  = F(13) + (size_t)bi * Lc;
      bp.m1_W1 = F(14) + (size_t)bi * Hc * Ec;  bp.m1_b1 = F(15) + (size_t)bi * Hc;
      bp.m1_W2 = F(16) + (size_t)bi * Hc * Hc;  bp.m1_b2 = F(17) + (size_t)bi * Hc;
      bp.m1_W3 = F(18) + (size_t)bi * Ec * Hc;  bp.m1_b3 = F(19) + (size_t)bi * Ec;
      bp.m2_Wq = F(20) + (size_t)bi * Ec * Ec;  bp.m2_bq = F(21) + (size_t)bi * Ec;
      bp.m2_Wk = F(22) + (size_t)bi * Ec * Ec;  bp.m2_bk = F(23) + (size_t)bi * Ec;
      bp.m2_Wv = F(24) + (size_t)bi * Ec * Ec;  bp.m2_bv = F(25) + (size_t)bi * Ec;
      bp.m2_Wo = F(26) + (size_t)bi * Ec * Ec;  bp.m2_bo = F(27) + (size_t)bi * Ec;
      bp.m3_Wq = F(28) + (size_t)bi * Ec * Ec;  bp.m3_bq = F(29) + (size_t)bi * Ec;
      bp.m3_Wk = F(30) + (size_t)bi * Ec * Ec;  bp.m3_bk = F(31) + (size_t)bi * Ec;
      bp.m3_Wv = F(32) + (size_t)bi * Ec * Ec;  bp.m3_bv = F(33) + (size_t)bi * Ec;
      bp.m3_Wo = F(34) + (size_t)bi * Ec * Ec;  bp.m3_bo = F(35) + (size_t)bi * Ec;
      bp.Wae   = F(36) + (size_t)bi * Ec * Ec;  bp.bae   = F(37) + (size_t)bi * Ec;
      bp.Wan   = F(38) + (size_t)bi * Lc * Lc;  bp.ban   = F(39) + (size_t)bi * Lc;
      bp.expl  = F(40) + (size_t)bi * 12;
      emb_update_kernel<<<Bc * Ac, 256, 0, stream>>>(cur, alt, bp);
      float* t = cur; cur = alt; alt = t;
    }
  }
  head_kernel<<<blks, 256, 0, stream>>>(cur, maskb, F(41), F(42), F(43), F(44),
                                        (float*)d_out);
}
